// TransformerBlockQuantum_65481071406748
// MI455X (gfx1250) — compile-verified, matrix operand order checked by reading, not run on hardware
//
#include <hip/hip_runtime.h>

typedef __attribute__((ext_vector_type(16))) _Float16 v16h;
typedef __attribute__((ext_vector_type(8)))  _Float16 v8h;
typedef __attribute__((ext_vector_type(8)))  float    v8f;
typedef __attribute__((ext_vector_type(4)))  float    v4f;

#define B_   4
#define S_   1024
#define E_   512
#define H_   64
#define DK_  8
#define FFN_ 2048
#define TOK_ (B_*S_)          // 4096 tokens
#define KC_  128              // K-chunk staged in LDS per GEMM iteration
#define SCALE_ 0.35355339059327373f   // 1/sqrt(DK)

// ---------------------------------------------------------------- helpers ---

static __device__ __forceinline__ v16h zero16() {
  v16h r;
#pragma unroll
  for (int i = 0; i < 16; ++i) r[i] = (_Float16)0.0f;
  return r;
}
static __device__ __forceinline__ v8f zero8f() {
  v8f r;
#pragma unroll
  for (int i = 0; i < 8; ++i) r[i] = 0.0f;
  return r;
}
static __device__ __forceinline__ v16h ld_pair(const _Float16* lo, const _Float16* hi) {
  const v8h a = *(const v8h*)lo;
  const v8h b = *(const v8h*)hi;
  v16h r;
#pragma unroll
  for (int i = 0; i < 8; ++i) { r[i] = a[i]; r[i + 8] = b[i]; }
  return r;
}
static __device__ __forceinline__ v8f wmma32(v16h a, v16h b, v8f c) {
  // D = A(16x32 f16) * B(32x16 f16) + C(16x16 f32)
  return __builtin_amdgcn_wmma_f32_16x16x32_f16(false, a, false, b, (short)0, c,
                                                false, false);
}

// Cross-lane xor reductions via VALU permlane (no LDS pipe / dscnt waits).
// permlane16 selector nibbles: lane i (within 16-group) reads lane nibble[i].
#define PERMXOR_F(x, lo, hi)                                                   \
  __builtin_bit_cast(float, __builtin_amdgcn_permlane16(                       \
      __builtin_bit_cast(int, (x)), __builtin_bit_cast(int, (x)), (lo), (hi),  \
      false, false))

static __device__ __forceinline__ float xor16_f(float x) {  // swap 16-lane halves
  int i = __builtin_bit_cast(int, x);
  i = __builtin_amdgcn_permlanex16(i, i, 0x76543210, 0xfedcba98, false, false);
  return __builtin_bit_cast(float, i);
}
static __device__ __forceinline__ float wave_sum(float s) {
  s += PERMXOR_F(s, 0x67452301, 0xefcdab89);  // xor 1
  s += PERMXOR_F(s, 0x54761032, 0xdcfe98ba);  // xor 2
  s += PERMXOR_F(s, 0x32107654, 0xba98fedc);  // xor 4
  s += PERMXOR_F(s, 0xfedcba98, 0x76543210);  // xor 8
  s += xor16_f(s);                            // xor 16
  return s;
}

// ---------------------------------------------- async global -> LDS staging ---
// One instruction per wave moves 32 lanes x 16B = 512B. 8 waves/block stage a
// full 16 x KC_ f16 A-tile (4 KB). Tracked by ASYNCcnt.
static __device__ __forceinline__ void async_copy_chunk(
    const _Float16* __restrict__ A, int K, int m0, int kc,
    _Float16* sbuf, int tid) {
  const int row = tid >> 4;          // 0..15
  const int seg = tid & 15;          // 0..15, 8 halves each
  const _Float16* g = A + (size_t)(m0 + row) * K + kc + seg * 8;
  _Float16* l = sbuf + row * KC_ + seg * 8;
  uint64_t ga = (uint64_t)(uintptr_t)g;
  uint32_t la = (uint32_t)(uintptr_t)l;   // low 32 bits of flat LDS addr = LDS offset
  asm volatile("global_load_async_to_lds_b128 %0, %1, off"
               :: "v"(la), "v"(ga) : "memory");
}
static __device__ __forceinline__ void wait_async0() {
  asm volatile("s_wait_asynccnt 0x0" ::: "memory");
}

// ------------------------------------------------- weight transpose (f32->f16T)
__global__ void k_transpose_h(const float* __restrict__ src, _Float16* __restrict__ dst,
                              int R, int C) {
  int idx = blockIdx.x * blockDim.x + threadIdx.x;
  if (idx >= R * C) return;
  int r = idx / C, c = idx - r * C;
  dst[(size_t)c * R + r] = (_Float16)src[idx];
}

// ------------------------------------------------------------- qkv encoding ---
__global__ void k_qkv_encode(const float* __restrict__ x, const float* __restrict__ phi,
                             _Float16* __restrict__ qkv, _Float16* __restrict__ qkvT) {
  int idx = blockIdx.x * blockDim.x + threadIdx.x;   // [0, 256*1024)
  int s  = idx & (S_ - 1);
  int bh = idx >> 10;                                // b*H + h
  int b = bh >> 6, h = bh & 63;
  const float* xp = x + ((size_t)(b * S_ + s)) * E_ + h * DK_;
  v4f x0 = *(const v4f*)xp;
  v4f x1 = *(const v4f*)(xp + 4);
  float q[8];
  q[0] = cosf(x0.x + phi[0]); q[1] = cosf(x0.y + phi[1]);
  q[2] = cosf(x0.z + phi[2]); q[3] = cosf(x0.w + phi[3]);
  q[4] = cosf(x1.x + phi[4]); q[5] = cosf(x1.y + phi[5]);
  q[6] = cosf(x1.z + phi[6]); q[7] = cosf(x1.w + phi[7]);
  v8h r;
#pragma unroll
  for (int d = 0; d < 8; ++d) r[d] = (_Float16)q[d];
  *(v8h*)(qkv + (size_t)bh * S_ * DK_ + (size_t)s * DK_) = r;
  _Float16* tp = qkvT + (size_t)bh * DK_ * S_ + s;
#pragma unroll
  for (int d = 0; d < 8; ++d) tp[(size_t)d * S_] = (_Float16)q[d];
}

// ---------------------------------------------------- fused flash attention ---
// One wave per (b,h,q-tile). S^T = K*Q^T via WMMA; online softmax in registers;
// out^T = V^T * P^T via a second WMMA whose B operand is the in-lane f16 repack
// of the exp'd scores (no cross-lane data movement needed).
__global__ void k_attention(const _Float16* __restrict__ qkv,
                            const _Float16* __restrict__ qkvT,
                            _Float16* __restrict__ outh) {
  const int lane = threadIdx.x & 31;
  const int wid  = blockIdx.x * (blockDim.x >> 5) + (threadIdx.x >> 5);
  const int qt = wid & 63;        // q-tile within head
  const int bh = wid >> 6;        // b*H + h
  const int n16 = lane & 15;
  const int hi  = lane >> 4;

  const _Float16* qbase = qkv  + (size_t)bh * S_ * DK_;
  const _Float16* tbase = qkvT + (size_t)bh * DK_ * S_;

  // B operand of score WMMA: Q^T (K-dim = features, 8 real of 32)
  v16h bq = zero16();
  {
    v8h t = *(const v8h*)(qbase + (size_t)(qt * 16 + n16) * DK_);
#pragma unroll
    for (int i = 0; i < 8; ++i) bq[i] = hi ? (_Float16)0.0f : t[i];
  }

  const _Float16* vrow = tbase + (size_t)(n16 & 7) * S_;  // V^T row (feature d)
  float mrun = -3.0e38f, lrun = 0.0f;
  v8f o = zero8f();

  for (int kt = 0; kt < S_ / 16; ++kt) {
    __builtin_prefetch(qbase + (size_t)(kt * 16 + 64) * DK_, 0, 0);
    // A operand: K-tile (16 keys x 32 features, 8 real). All lanes issue a
    // valid load; the zero padding is applied with register selects.
    v16h ak = zero16();
    {
      v8h t = *(const v8h*)(qbase + (size_t)(kt * 16 + n16) * DK_);
#pragma unroll
      for (int i = 0; i < 8; ++i) ak[i] = hi ? (_Float16)0.0f : t[i];
    }
    v8f c = wmma32(ak, bq, zero8f());   // c[v]: key = v+8*hi, query = n16

    // online softmax over keys (per query column)
    float tm = -3.0e38f;
#pragma unroll
    for (int v = 0; v < 8; ++v) tm = fmaxf(tm, c[v]);
    tm *= SCALE_;
    tm = fmaxf(tm, xor16_f(tm));
    float mnew = fmaxf(mrun, tm);
    float corr = __expf(mrun - mnew);
    float tsum = 0.0f;
    v16h pb = zero16();                 // B operand of PV WMMA (keys 16..31 = 0)
#pragma unroll
    for (int v = 0; v < 8; ++v) {
      float e = __expf(c[v] * SCALE_ - mnew);
      tsum += e;
      pb[v] = (_Float16)e;
    }
    tsum += xor16_f(tsum);
    lrun = lrun * corr + tsum;
    mrun = mnew;
#pragma unroll
    for (int v = 0; v < 8; ++v) o[v] *= corr;

    // A operand: V^T (rows = features, 8 real of 16; K = keys, 16 real of 32)
    v16h av = zero16();
    {
      v8h t = *(const v8h*)(vrow + kt * 16 + hi * 8);
#pragma unroll
      for (int i = 0; i < 8; ++i) av[i] = (n16 < 8) ? t[i] : (_Float16)0.0f;
    }
    o = wmma32(av, pb, o);              // o[v]: feature = v+8*hi, query = n16
  }

  if (hi == 0) {                        // real features 0..7 live in the low half
    float inv = 1.0f / lrun;
    int b = bh >> 6, h = bh & 63;
    v8h r;
#pragma unroll
    for (int v = 0; v < 8; ++v) r[v] = (_Float16)(o[v] * inv);
    *(v8h*)(outh + ((size_t)(b * S_ + qt * 16 + n16)) * E_ + h * DK_) = r;
  }
}

// ----------------------------------------------- LDS-staged WMMA GEMM core ---
// A-tile (16 rows, shared by all 8 waves of the block) is staged in LDS via
// async copies, double-buffered. BT rows (per-wave) stream from global.
static __device__ __forceinline__ v8f wmma_chunk_lds(
    const _Float16* sbuf, const _Float16* br, int kc, int lane, v8f c) {
  const int n16 = lane & 15;
  const int ko  = (lane >> 4) * 8;
  const _Float16* ar = sbuf + n16 * KC_;
#pragma unroll
  for (int kl = 0; kl < KC_; kl += 32) {
    v16h a = ld_pair(ar + kl + ko, ar + kl + 16 + ko);            // ds_load_b128
    v16h b = ld_pair(br + kc + kl + ko, br + kc + kl + 16 + ko);  // global b128
    c = wmma32(a, b, c);
  }
  return c;
}

// GEMM1: tmp1 = attn_out @ Wc + bc + x   (residual fused)
__global__ void k_gemm_combine(const _Float16* __restrict__ Ah,
                               const _Float16* __restrict__ WcT,
                               const float* __restrict__ bc,
                               const float* __restrict__ x,
                               float* __restrict__ tmp1) {
  __shared__ _Float16 sA[2][16 * KC_];
  const int tid  = threadIdx.x;
  const int lane = tid & 31;
  const int wid  = blockIdx.x * 8 + (tid >> 5);
  const int NT = E_ / 16;                 // 32: all 8 waves share mt
  const int mt = wid / NT, nt = wid - mt * NT;
  const int m0 = mt * 16, n0 = nt * 16;
  const int n16 = lane & 15, hi = lane >> 4;
  const _Float16* br = WcT + (size_t)(n0 + n16) * E_;

  async_copy_chunk(Ah, E_, m0, 0, sA[0], tid);
  v8f c = zero8f();
  for (int ch = 0; ch < E_ / KC_; ++ch) {
    wait_async0();
    __syncthreads();
    if (ch + 1 < E_ / KC_)
      async_copy_chunk(Ah, E_, m0, (ch + 1) * KC_, sA[(ch + 1) & 1], tid);
    c = wmma_chunk_lds(sA[ch & 1], br, ch * KC_, lane, c);
    __syncthreads();
  }
  const int col = n0 + n16;
  const float bcv = bc[col];
#pragma unroll
  for (int v = 0; v < 8; ++v) {
    int row = m0 + v + 8 * hi;
    size_t idx = (size_t)row * E_ + col;
    tmp1[idx] = c[v] + bcv + x[idx];
  }
}

// LN1: x1 = layernorm(tmp1)*g1+be1 ; also qout = cos(x1[:,:8])*cos(theta) (f16)
__global__ void k_ln1(const float* __restrict__ tmp1, const float* __restrict__ g,
                      const float* __restrict__ be, const float* __restrict__ theta,
                      float* __restrict__ x1, _Float16* __restrict__ qouth) {
  const int lane = threadIdx.x & 31;
  const int tok  = blockIdx.x * (blockDim.x >> 5) + (threadIdx.x >> 5);
  const v4f* p = (const v4f*)(tmp1 + (size_t)tok * E_ + lane * 16);
  float vb[16];
#pragma unroll
  for (int j = 0; j < 4; ++j) {
    v4f t = p[j];
    vb[j*4+0] = t.x; vb[j*4+1] = t.y; vb[j*4+2] = t.z; vb[j*4+3] = t.w;
  }
  float s = 0.0f, sq = 0.0f;
#pragma unroll
  for (int i = 0; i < 16; ++i) { s += vb[i]; sq += vb[i] * vb[i]; }
  s  = wave_sum(s);
  sq = wave_sum(sq);
  const float mu = s * (1.0f / E_);
  const float rs = rsqrtf(sq * (1.0f / E_) - mu * mu + 1e-5f);
  float yb[16];
  v4f* po = (v4f*)(x1 + (size_t)tok * E_ + lane * 16);
#pragma unroll
  for (int j = 0; j < 4; ++j) {
    v4f t;
    int c0 = lane * 16 + j * 4;
    t.x = (vb[j*4+0] - mu) * rs * g[c0+0] + be[c0+0];
    t.y = (vb[j*4+1] - mu) * rs * g[c0+1] + be[c0+1];
    t.z = (vb[j*4+2] - mu) * rs * g[c0+2] + be[c0+2];
    t.w = (vb[j*4+3] - mu) * rs * g[c0+3] + be[c0+3];
    yb[j*4+0] = t.x; yb[j*4+1] = t.y; yb[j*4+2] = t.z; yb[j*4+3] = t.w;
    po[j] = t;
  }
  if (lane == 0) {                      // channels 0..7 live in lane 0
    v8h r;
#pragma unroll
    for (int j = 0; j < 8; ++j) r[j] = (_Float16)(cosf(yb[j]) * cosf(theta[j]));
    *(v8h*)(qouth + (size_t)tok * DK_) = r;
  }
}

// GEMM2: h = relu(qout @ W1 + b1) ; K = 8 zero-padded to 32, single WMMA/tile
__global__ void k_gemm_ffn1(const _Float16* __restrict__ qouth,
                            const _Float16* __restrict__ W1T,
                            const float* __restrict__ b1,
                            _Float16* __restrict__ hh) {
  const int lane = threadIdx.x & 31;
  const int wid  = blockIdx.x * (blockDim.x >> 5) + (threadIdx.x >> 5);
  const int NT = FFN_ / 16;
  const int mt = wid / NT, nt = wid - mt * NT;
  const int n16 = lane & 15, hi = lane >> 4;
  v16h a = zero16(), b = zero16();
  if (hi == 0) {
    v8h ta = *(const v8h*)(qouth + (size_t)(mt * 16 + n16) * DK_);
    v8h tb = *(const v8h*)(W1T   + (size_t)(nt * 16 + n16) * DK_);
#pragma unroll
    for (int i = 0; i < 8; ++i) { a[i] = ta[i]; b[i] = tb[i]; }
  }
  v8f c = wmma32(a, b, zero8f());
  const int col = nt * 16 + n16;
  const float bv = b1[col];
#pragma unroll
  for (int v = 0; v < 8; ++v) {
    int row = mt * 16 + v + 8 * hi;
    hh[(size_t)row * FFN_ + col] = (_Float16)fmaxf(c[v] + bv, 0.0f);
  }
}

// GEMM3: ffn = h @ W2 + b2   (LDS-staged A, K = 2048)
__global__ void k_gemm_ffn2(const _Float16* __restrict__ hh,
                            const _Float16* __restrict__ W2T,
                            const float* __restrict__ b2,
                            float* __restrict__ ffn) {
  __shared__ _Float16 sA[2][16 * KC_];
  const int tid  = threadIdx.x;
  const int lane = tid & 31;
  const int wid  = blockIdx.x * 8 + (tid >> 5);
  const int NT = E_ / 16;                 // 32: all 8 waves share mt
  const int mt = wid / NT, nt = wid - mt * NT;
  const int m0 = mt * 16, n0 = nt * 16;
  const int n16 = lane & 15, hi = lane >> 4;
  const _Float16* br = W2T + (size_t)(n0 + n16) * FFN_;

  async_copy_chunk(hh, FFN_, m0, 0, sA[0], tid);
  v8f c = zero8f();
  for (int ch = 0; ch < FFN_ / KC_; ++ch) {
    wait_async0();
    __syncthreads();
    if (ch + 1 < FFN_ / KC_)
      async_copy_chunk(hh, FFN_, m0, (ch + 1) * KC_, sA[(ch + 1) & 1], tid);
    c = wmma_chunk_lds(sA[ch & 1], br, ch * KC_, lane, c);
    __syncthreads();
  }
  const int col = n0 + n16;
  const float bv = b2[col];
#pragma unroll
  for (int v = 0; v < 8; ++v) {
    int row = m0 + v + 8 * hi;
    ffn[(size_t)row * E_ + col] = c[v] + bv;
  }
}

// LN2: out = layernorm(x1 + ffn)*g2+be2
__global__ void k_ln2(const float* __restrict__ x1, const float* __restrict__ ffn,
                      const float* __restrict__ g, const float* __restrict__ be,
                      float* __restrict__ out) {
  const int lane = threadIdx.x & 31;
  const int tok  = blockIdx.x * (blockDim.x >> 5) + (threadIdx.x >> 5);
  const v4f* pa = (const v4f*)(x1  + (size_t)tok * E_ + lane * 16);
  const v4f* pb = (const v4f*)(ffn + (size_t)tok * E_ + lane * 16);
  float vb[16];
#pragma unroll
  for (int j = 0; j < 4; ++j) {
    v4f ta = pa[j], tb = pb[j];
    vb[j*4+0] = ta.x + tb.x; vb[j*4+1] = ta.y + tb.y;
    vb[j*4+2] = ta.z + tb.z; vb[j*4+3] = ta.w + tb.w;
  }
  float s = 0.0f, sq = 0.0f;
#pragma unroll
  for (int i = 0; i < 16; ++i) { s += vb[i]; sq += vb[i] * vb[i]; }
  s  = wave_sum(s);
  sq = wave_sum(sq);
  const float mu = s * (1.0f / E_);
  const float rs = rsqrtf(sq * (1.0f / E_) - mu * mu + 1e-5f);
  v4f* po = (v4f*)(out + (size_t)tok * E_ + lane * 16);
#pragma unroll
  for (int j = 0; j < 4; ++j) {
    int c0 = lane * 16 + j * 4;
    v4f t;
    t.x = (vb[j*4+0] - mu) * rs * g[c0+0] + be[c0+0];
    t.y = (vb[j*4+1] - mu) * rs * g[c0+1] + be[c0+1];
    t.z = (vb[j*4+2] - mu) * rs * g[c0+2] + be[c0+2];
    t.w = (vb[j*4+3] - mu) * rs * g[c0+3] + be[c0+3];
    po[j] = t;
  }
}

// ------------------------------------------------------------------ launch ---
extern "C" void kernel_launch(void* const* d_in, const int* in_sizes, int n_in,
                              void* d_out, int out_size, void* d_ws, size_t ws_size,
                              hipStream_t stream) {
  (void)in_sizes; (void)n_in; (void)out_size; (void)ws_size;
  const float* x     = (const float*)d_in[0];
  const float* phi   = (const float*)d_in[1];
  const float* Wc    = (const float*)d_in[2];
  const float* bc    = (const float*)d_in[3];
  const float* theta = (const float*)d_in[4];
  const float* W1    = (const float*)d_in[5];
  const float* b1    = (const float*)d_in[6];
  const float* W2    = (const float*)d_in[7];
  const float* b2    = (const float*)d_in[8];
  const float* g1    = (const float*)d_in[9];
  const float* be1   = (const float*)d_in[10];
  const float* g2    = (const float*)d_in[11];
  const float* be2   = (const float*)d_in[12];
  float* out = (float*)d_out;

  char* w = (char*)d_ws;
  size_t off = 0;
  auto take = [&](size_t bytes) -> char* {
    char* p = w + off;
    off = (off + bytes + 255) & ~(size_t)255;
    return p;
  };
  _Float16* qkv_h  = (_Float16*)take((size_t)B_*H_*S_*DK_ * 2);   // 4 MB
  _Float16* qkvT_h = (_Float16*)take((size_t)B_*H_*DK_*S_ * 2);   // 4 MB
  _Float16* attn_h = (_Float16*)take((size_t)TOK_*E_ * 2);        // 4 MB
  _Float16* WcT_h  = (_Float16*)take((size_t)E_*E_ * 2);          // 512 KB
  _Float16* W1T_h  = (_Float16*)take((size_t)DK_*FFN_ * 2);       // 32 KB
  _Float16* W2T_h  = (_Float16*)take((size_t)FFN_*E_ * 2);        // 2 MB
  float*    tmp1   = (float*)   take((size_t)TOK_*E_ * 4);        // 8 MB
  float*    x1     = (float*)   take((size_t)TOK_*E_ * 4);        // 8 MB
  _Float16* qout_h = (_Float16*)take((size_t)TOK_*DK_ * 2);       // 64 KB
  _Float16* h_h    = (_Float16*)take((size_t)TOK_*FFN_ * 2);      // 16 MB
  float*    ffn    = (float*)   take((size_t)TOK_*E_ * 4);        // 8 MB

  const int T = 256;   // 8 waves per block

  // weight transposes (f32 -> f16, [N][K] row-major)
  k_transpose_h<<<(E_*E_   + T-1)/T, T, 0, stream>>>(Wc, WcT_h, E_,   E_);
  k_transpose_h<<<(DK_*FFN_+ T-1)/T, T, 0, stream>>>(W1, W1T_h, DK_,  FFN_);
  k_transpose_h<<<(FFN_*E_ + T-1)/T, T, 0, stream>>>(W2, W2T_h, FFN_, E_);

  // qkv encoding
  k_qkv_encode<<<(B_*H_*S_)/T, T, 0, stream>>>(x, phi, qkv_h, qkvT_h);

  // fused flash attention: B*H*(S/16) = 16384 waves
  k_attention<<<(B_*H_*(S_/16))/8, T, 0, stream>>>(qkv_h, qkvT_h, attn_h);

  // combine heads + residual (LDS-staged WMMA GEMM)
  k_gemm_combine<<<((TOK_/16)*(E_/16))/8, T, 0, stream>>>(attn_h, WcT_h, bc, x, tmp1);

  // layernorm 1 + quantum FFN encoding
  k_ln1<<<TOK_/8, T, 0, stream>>>(tmp1, g1, be1, theta, x1, qout_h);

  // FFN
  k_gemm_ffn1<<<((TOK_/16)*(FFN_/16))/8, T, 0, stream>>>(qout_h, W1T_h, b1, h_h);
  k_gemm_ffn2<<<((TOK_/16)*(E_/16))/8,  T, 0, stream>>>(h_h, W2T_h, b2, ffn);

  // layernorm 2 -> output
  k_ln2<<<TOK_/8, T, 0, stream>>>(x1, ffn, g2, be2, out);
}
